// DeformableKPConvLayer_48034914238863
// MI455X (gfx1250) — matrix operand order repaired and checked by reading, not compiled, not run on hardware
//
#include <hip/hip_runtime.h>
#include <hip/hip_bf16.h>

// DeformableKPConvLayer for MI455X (gfx1250), wave32 + WMMA f16.
//  - prep kernel: one-time f32->f16 conversion; weight/dweight reordered to
//    [f*16+kpt][o] rows so the wf LDS buffer can be f-major (contiguous D stores).
//  - main kernel: one block = 16 queries, 256 threads = 8 waves.
//    Phases B/D use V_WMMA_F16_16X16X32_F16 (f16 D -> zero convert, 1 ds_store_b128
//    per tile); phases C/E accumulate in f32.

typedef __attribute__((ext_vector_type(16))) _Float16 v16h;
typedef __attribute__((ext_vector_type(8)))  _Float16 v8h;
typedef __attribute__((ext_vector_type(8)))  float    v8f;

#define NQ      16384
#define NS      16384
#define NN      32
#define FIN     128
#define FOUT    256
#define KPTS    15
#define KP16    16            // padded kernel-point dim
#define KBIG    (FIN * KP16)  // 2048 = flattened contraction dim (f*16 + kpt)
#define OFFD    42
#define OFFP    48            // padded offset dim (3 x 16-wide N tiles)
#define QPB     16            // queries per block
#define INV_EXTENT 0.5f       // 1 / EXTENT, EXTENT = 2.0
#define SHADOW  1.0e6f        // pad kernel point -> relu influence == 0 naturally

// Workspace layout (_Float16 element offsets):
//   feats_h : [16384][128]          (4 MB)
//   wgt2_h  : [2048][256]  row = f*16+kpt, kpt==15 rows zero   (1 MB)
//   dwe2_h  : [2048][48]   row = f*16+kpt, padded cols zero    (192 KB)
#define WS_FEATS 0
#define WS_WGT   (NS * FIN)
#define WS_DWE   (WS_WGT + KBIG * FOUT)

#if defined(__has_builtin)
#if __has_builtin(__builtin_amdgcn_wmma_f16_16x16x32_f16)
#define HAVE_WMMA_F16D 1
#endif
#endif
#ifndef HAVE_WMMA_F16D
#define HAVE_WMMA_F16D 0
#endif

// D = A x B with f16 result tile (per-lane v8h = rows mb..mb+7 of one column).
__device__ __forceinline__ v8h wmma_f16d(v16h a, v16h b) {
#if HAVE_WMMA_F16D
  v8h c = {};
  return __builtin_amdgcn_wmma_f16_16x16x32_f16(false, a, false, b, (short)0, c,
                                                false, false);
#else
  v8f c = {};
  c = __builtin_amdgcn_wmma_f32_16x16x32_f16(false, a, false, b, (short)0, c,
                                             false, false);
  v8h d;
#pragma unroll
  for (int r = 0; r < 8; ++r) d[r] = (_Float16)c[r];
  return d;
#endif
}

// ---------------------------------------------------------------------------
__global__ __launch_bounds__(256)
void prep_f16_kernel(const float* __restrict__ feats,
                     const float* __restrict__ weight,
                     const float* __restrict__ dweight,
                     _Float16*    __restrict__ ws) {
  const size_t i = (size_t)blockIdx.x * 256 + threadIdx.x;
  if (i < (size_t)NS * FIN) ws[WS_FEATS + i] = (_Float16)feats[i];
  if (i < (size_t)KBIG * FOUT) {           // wgt2[(f*16+kpt)][o] = weight[kpt][f][o]
    const int r = (int)(i >> 8), o = (int)(i & 255);
    const int f = r >> 4, kpt = r & 15;
    ws[WS_WGT + i] = (kpt < KPTS)
        ? (_Float16)weight[((size_t)kpt * FIN + f) * FOUT + o] : (_Float16)0.0f;
  }
  if (i < (size_t)KBIG * OFFP) {           // dwe2[(f*16+kpt)][o] = dweight[kpt][f][o]
    const int r = (int)(i / OFFP), o = (int)(i - (size_t)r * OFFP);
    const int f = r >> 4, kpt = r & 15;
    ws[WS_DWE + i] = (kpt < KPTS && o < OFFD)
        ? (_Float16)dweight[((size_t)kpt * FIN + f) * OFFD + o] : (_Float16)0.0f;
  }
}

// Load 16 f16 A-matrix elements for one lane from the LDS wf buffer.
// A is [M=16 queries][K'' = f*16 + kpt] (flat 2048); 16-bit A layout:
//   lanes 0-15 : elements 0-7 -> K''=koff+0..7,  8-15 -> K''=koff+16..23
//   lanes 16-31: elements 0-7 -> K''=koff+8..15, 8-15 -> K''=koff+24..31
__device__ __forceinline__ v16h wf_load_A(const _Float16* wq, int koff, int khalf) {
  const _Float16* p = wq + koff + khalf * 8;
  v8h lo = *(const v8h*)(p);
  v8h hi = *(const v8h*)(p + 16);
  return __builtin_shufflevector(lo, hi, 0,1,2,3,4,5,6,7,8,9,10,11,12,13,14,15);
}

// ---------------------------------------------------------------------------
__global__ __launch_bounds__(256)
void kpconv_deform_kernel(const float* __restrict__ qpts,
                          const float* __restrict__ spts,
                          const int*   __restrict__ nbrs,
                          const float* __restrict__ kpoints,
                          const float* __restrict__ bias,
                          const _Float16* __restrict__ ws,
                          float*       __restrict__ out) {
  __shared__ __align__(16) float    s_rel[QPB][NN][3];
  __shared__ int                    s_nbr[QPB][NN];
  __shared__ __align__(16) _Float16 s_wf[QPB][FIN][KP16];  // 64 KB, f-major
  __shared__ float                  s_kp[16][3];
  __shared__ float                  s_defkp[QPB][16][3];
  __shared__ float                  s_off[QPB][OFFP];
  __shared__ float                  s_bias[OFFP];

  const _Float16* __restrict__ feats_h = ws + WS_FEATS;
  const _Float16* __restrict__ wgt_h   = ws + WS_WGT;
  const _Float16* __restrict__ dwe_h   = ws + WS_DWE;

  const int tid   = threadIdx.x;
  const int wave  = tid >> 5;
  const int lane  = tid & 31;
  const int m     = lane & 15;
  const int khalf = lane >> 4;
  const int mb    = khalf * 8;
  const int qbase = blockIdx.x * QPB;

  // ---------- Phase A: stage kernel points, bias, neighbor ids, rel ---------
  if (tid < KPTS * 3) ((float*)s_kp)[tid] = kpoints[tid];
  if (tid >= KPTS * 3 && tid < 48) ((float*)s_kp)[tid] = SHADOW;  // pad row 15
  if (tid < OFFP) s_bias[tid] = (tid < OFFD) ? bias[tid] : 0.0f;
  // Warm L2 with the big output-GEMM weight while phases A-D run.
  __builtin_prefetch(wgt_h + (size_t)tid * 512, 0, 1);

#pragma unroll
  for (int i = tid; i < QPB * NN; i += 256) {
    const int q  = i >> 5;
    const int nb = i & 31;
    const int gq = qbase + q;
    const int idx = nbrs[gq * NN + nb];
    s_nbr[q][nb] = idx;
    s_rel[q][nb][0] = spts[idx * 3 + 0] - qpts[gq * 3 + 0];
    s_rel[q][nb][1] = spts[idx * 3 + 1] - qpts[gq * 3 + 1];
    s_rel[q][nb][2] = spts[idx * 3 + 2] - qpts[gq * 3 + 2];
  }
  __syncthreads();

  // ---------- Phase B: rigid influences -> wf0 = w0^T @ nfeat (per query) ----
#pragma unroll
  for (int qi = 0; qi < 2; ++qi) {
    const int q = wave * 2 + qi;
    v16h a;
    {
      const float kx = s_kp[m][0], ky = s_kp[m][1], kz = s_kp[m][2];
#pragma unroll
      for (int e = 0; e < 16; ++e) {
        const int nb = (e & 7) + khalf * 8 + ((e & 8) ? 16 : 0);
        const float dx = s_rel[q][nb][0] - kx;
        const float dy = s_rel[q][nb][1] - ky;
        const float dz = s_rel[q][nb][2] - kz;
        const float d = __builtin_amdgcn_sqrtf(dx * dx + dy * dy + dz * dz);
        a[e] = (_Float16)fmaxf(1.0f - d * INV_EXTENT, 0.0f);
      }
    }
    const _Float16* frow = feats_h + (size_t)s_nbr[q][lane] * FIN;  // lane = nbr
#pragma unroll
    for (int ft = 0; ft < 8; ++ft) {
      v16h b;
#pragma unroll
      for (int e = 0; e < 16; ++e) {
        const int nb = (e & 7) + khalf * 8 + ((e & 8) ? 16 : 0);
        b[e] = feats_h[(size_t)s_nbr[q][nb] * FIN + ft * 16 + m];
      }
      v8h  d = wmma_f16d(a, b);                    // rows mb..mb+7, col f
      *(v8h*)(&s_wf[q][ft * 16 + m][mb]) = d;      // one ds_store_b128
    }
  }
  __syncthreads();

  // ---------- Phase C: offsets GEMM  [16 q x 2048] x [2048 x 48] -------------
  if (wave < 3) {
    const int o0 = wave * 16;
    const _Float16* wq = &s_wf[m][0][0];  // A row = query m, flat K'' = f*16+kpt
    v8f c = {};
#pragma unroll 4
    for (int kb = 0; kb < KBIG / 32; ++kb) {
      v16h a = wf_load_A(wq, kb * 32, khalf);
      v16h b;
#pragma unroll
      for (int e = 0; e < 16; ++e)
        b[e] = dwe_h[((size_t)kb * 32 + (e & 7) + khalf * 8 + ((e & 8) ? 16 : 0)) * OFFP + o0 + m];
      c = __builtin_amdgcn_wmma_f32_16x16x32_f16(false, a, false, b, (short)0, c,
                                                 false, false);
    }
    const int o = o0 + m;
    const float bo = s_bias[o];
#pragma unroll
    for (int r = 0; r < 8; ++r) s_off[mb + r][o] = c[r] + bo;
  }
  __syncthreads();

  // ---------- deformed kernel points (row 15 = shadow point) -----------------
  for (int i = tid; i < QPB * 16 * 3; i += 256) {
    const int q = i / 48, rem = i % 48, kp = rem / 3, d = rem - kp * 3;
    float v = SHADOW;
    if (kp < KPTS) {
      const float off = (kp == 0) ? 0.0f : s_off[q][(kp - 1) * 3 + d];
      v = s_kp[kp][d] + off;     // KP_EXTENT_CFG == 1.0
    }
    s_defkp[q][kp][d] = v;
  }
  __syncthreads();

  // ---------- Phase D: deformed influences -> wf1 (overwrite s_wf) -----------
#pragma unroll
  for (int qi = 0; qi < 2; ++qi) {
    const int q = wave * 2 + qi;
    v16h a;
    {
      const float kx = s_defkp[q][m][0];
      const float ky = s_defkp[q][m][1];
      const float kz = s_defkp[q][m][2];
#pragma unroll
      for (int e = 0; e < 16; ++e) {
        const int nb = (e & 7) + khalf * 8 + ((e & 8) ? 16 : 0);
        const float dx = s_rel[q][nb][0] - kx;
        const float dy = s_rel[q][nb][1] - ky;
        const float dz = s_rel[q][nb][2] - kz;
        const float d = __builtin_amdgcn_sqrtf(dx * dx + dy * dy + dz * dz);
        a[e] = (_Float16)fmaxf(1.0f - d * INV_EXTENT, 0.0f);
      }
    }
    const _Float16* frow = feats_h + (size_t)s_nbr[q][lane] * FIN;
#pragma unroll
    for (int ft = 0; ft < 8; ++ft) {
      v16h b;
#pragma unroll
      for (int e = 0; e < 16; ++e) {
        const int nb = (e & 7) + khalf * 8 + ((e & 8) ? 16 : 0);
        b[e] = feats_h[(size_t)s_nbr[q][nb] * FIN + ft * 16 + m];
      }
      v8h  d = wmma_f16d(a, b);
      *(v8h*)(&s_wf[q][ft * 16 + m][mb]) = d;
    }
  }
  __syncthreads();

  // ---------- Phase E: output GEMM [16 q x 2048] x [2048 x 256] --------------
#pragma unroll
  for (int t = 0; t < 2; ++t) {
    const int o0 = (wave * 2 + t) * 16;
    const _Float16* wq = &s_wf[m][0][0];
    v8f c = {};
#pragma unroll 4
    for (int kb = 0; kb < KBIG / 32; ++kb) {
      v16h a = wf_load_A(wq, kb * 32, khalf);
      v16h b;
#pragma unroll
      for (int e = 0; e < 16; ++e)
        b[e] = wgt_h[((size_t)kb * 32 + (e & 7) + khalf * 8 + ((e & 8) ? 16 : 0)) * FOUT + o0 + m];
      c = __builtin_amdgcn_wmma_f32_16x16x32_f16(false, a, false, b, (short)0, c,
                                                 false, false);
    }
    const int o = o0 + m;
#pragma unroll
    for (int r = 0; r < 8; ++r)
      out[(size_t)(qbase + mb + r) * FOUT + o] = c[r];
  }
}

// ---------------------------------------------------------------------------
extern "C" void kernel_launch(void* const* d_in, const int* in_sizes, int n_in,
                              void* d_out, int out_size, void* d_ws, size_t ws_size,
                              hipStream_t stream) {
  const float* qpts    = (const float*)d_in[0];  // [16384,3]
  const float* spts    = (const float*)d_in[1];  // [16384,3]
  const int*   nbrs    = (const int*)  d_in[2];  // [16384,32]
  const float* feats   = (const float*)d_in[3];  // [16384,128]
  const float* kpoints = (const float*)d_in[4];  // [15,3]
  const float* weight  = (const float*)d_in[5];  // [15,128,256]
  const float* dweight = (const float*)d_in[6];  // [15,128,42]
  const float* bias    = (const float*)d_in[7];  // [42]
  float* out = (float*)d_out;                    // [16384,256]
  _Float16* ws = (_Float16*)d_ws;

  // One-time f16 conversion + weight reordering (deterministic every launch).
  const size_t prep_elems = (size_t)NS * FIN;    // largest segment
  const int prep_blocks = (int)((prep_elems + 255) / 256);
  prep_f16_kernel<<<prep_blocks, 256, 0, stream>>>(feats, weight, dweight, ws);

  const int blocks = NQ / QPB;  // 1024
  kpconv_deform_kernel<<<blocks, 256, 0, stream>>>(qpts, spts, nbrs, kpoints, bias,
                                                   ws, out);
}
